// UnaryLinear_24515673326112
// MI455X (gfx1250) — compile-verified, matrix operand order checked by reading, not run on hardware
//
#include <hip/hip_runtime.h>
#include <hip/hip_bf16.h>

// UnaryLinear (stochastic-computing linear, batch=1, 4096x4096).
// Memory-bound: streams buf_wght (64MB f32) + rng_wght_idx + rng_wght_idx_inv
// (64MB i32 each) exactly once => ~192MB => ~8.2us floor @ 23.3TB/s.
// Each wave computes a 16-row tile; 0/1 bits are packed as f16 into the CDNA5
// WMMA A-matrix layout and reduced with v_wmma_f32_16x16x32_f16 against an
// all-ones B (every accumulator column == row sum), so the matrix unit does
// the accumulate + cross-lane reduction. NT hints + prefetch on the streams.

typedef __attribute__((ext_vector_type(16))) _Float16 v16h;
typedef __attribute__((ext_vector_type(8)))  float    v8f;
typedef __attribute__((ext_vector_type(4)))  float    f32x4;
typedef __attribute__((ext_vector_type(4)))  int      i32x4;

#define IN_F  4096
#define OUT_F 4096
#define ROWS_PER_BLOCK 16
#define THREADS 512
#define NWAVES (THREADS / 32)          // 16
#define K_PER_WAVE (IN_F / NWAVES)     // 256
#define KSTEPS (K_PER_WAVE / 32)       // 8 wmma steps per wave

__global__ __launch_bounds__(THREADS)
void unary_linear_wmma(const float* __restrict__ x,
                       const float* __restrict__ wght,
                       const float* __restrict__ bias,
                       const float* __restrict__ rng,
                       const float* __restrict__ acc,
                       const float* __restrict__ bound,
                       const int*   __restrict__ idxw,
                       const int*   __restrict__ idxb,
                       const int*   __restrict__ idxinv,
                       float* __restrict__ out)
{
    __shared__ __align__(16) float s_x[IN_F];     // 16 KB
    __shared__ float s_rng[256];                  // 1 KB Sobol table
    __shared__ float s_part[ROWS_PER_BLOCK];      // per-row partial sums

    const int tid = threadIdx.x;
    for (int i = tid; i < IN_F; i += THREADS) s_x[i] = x[i];
    if (tid < 256) s_rng[tid] = rng[tid];
    if (tid < ROWS_PER_BLOCK) s_part[tid] = 0.0f;
    __syncthreads();

    const int wave = tid >> 5;
    const int lane = tid & 31;
    const int l16  = lane & 15;      // row within tile (A-matrix M)
    const int h    = lane >> 4;      // selects k-runs 8h.. and 16+8h..
    const int rowBase = blockIdx.x * ROWS_PER_BLOCK;
    const int row = rowBase + l16;

    const float* wrow = wght   + (size_t)row * IN_F;
    const int*   irow = idxw   + (size_t)row * IN_F;
    const int*   vrow = idxinv + (size_t)row * IN_F;

    const v16h ones = {
        (_Float16)1.0f, (_Float16)1.0f, (_Float16)1.0f, (_Float16)1.0f,
        (_Float16)1.0f, (_Float16)1.0f, (_Float16)1.0f, (_Float16)1.0f,
        (_Float16)1.0f, (_Float16)1.0f, (_Float16)1.0f, (_Float16)1.0f,
        (_Float16)1.0f, (_Float16)1.0f, (_Float16)1.0f, (_Float16)1.0f };

    v8f c = {};                       // D[m,n] accumulates row sums (all n equal)
    const int k0 = wave * K_PER_WAVE;

    for (int s = 0; s < KSTEPS; ++s) {
        const int kb = k0 + s * 32 + 8 * h;   // first 8-run; second at kb+16

        // Read-once streams: non-temporal b128 loads, prefetch next step.
        f32x4 w0 = __builtin_nontemporal_load((const f32x4*)(wrow + kb));
        f32x4 w1 = __builtin_nontemporal_load((const f32x4*)(wrow + kb + 4));
        f32x4 w2 = __builtin_nontemporal_load((const f32x4*)(wrow + kb + 16));
        f32x4 w3 = __builtin_nontemporal_load((const f32x4*)(wrow + kb + 20));
        i32x4 i0 = __builtin_nontemporal_load((const i32x4*)(irow + kb));
        i32x4 i1 = __builtin_nontemporal_load((const i32x4*)(irow + kb + 4));
        i32x4 i2 = __builtin_nontemporal_load((const i32x4*)(irow + kb + 16));
        i32x4 i3 = __builtin_nontemporal_load((const i32x4*)(irow + kb + 20));
        i32x4 v0 = __builtin_nontemporal_load((const i32x4*)(vrow + kb));
        i32x4 v1 = __builtin_nontemporal_load((const i32x4*)(vrow + kb + 4));
        i32x4 v2 = __builtin_nontemporal_load((const i32x4*)(vrow + kb + 16));
        i32x4 v3 = __builtin_nontemporal_load((const i32x4*)(vrow + kb + 20));
        if (s + 1 < KSTEPS) {
            __builtin_prefetch(wrow + kb + 32, 0, 0);
            __builtin_prefetch(irow + kb + 32, 0, 0);
            __builtin_prefetch(vrow + kb + 32, 0, 0);
        }
        f32x4 x0 = *(const f32x4*)(s_x + kb);
        f32x4 x1 = *(const f32x4*)(s_x + kb + 4);
        f32x4 x2 = *(const f32x4*)(s_x + kb + 16);
        f32x4 x3 = *(const f32x4*)(s_x + kb + 20);

        // e = x ? (w >= rng[idx]) : (w < rng[idx_inv])  ->  (xb == ge)
        v16h a;
        auto pack4 = [&](int base, f32x4 wv, i32x4 iv, i32x4 vv, f32x4 xv) {
#pragma unroll
            for (int j = 0; j < 4; ++j) {
                const bool  xb  = xv[j] >= 0.5f;
                const int   sel = (xb ? iv[j] : vv[j]) & 255;
                const float r   = s_rng[sel];
                const bool  ge  = wv[j] >= r;
                a[base + j] = (xb == ge) ? (_Float16)1.0f : (_Float16)0.0f;
            }
        };
        pack4(0,  w0, i0, v0, x0);   // k = kb .. kb+3
        pack4(4,  w1, i1, v1, x1);   // k = kb+4 .. kb+7
        pack4(8,  w2, i2, v2, x2);   // k = kb+16 .. kb+19
        pack4(12, w3, i3, v3, x3);   // k = kb+20 .. kb+23

        // D += A(16x32 bits) x B(ones): every column of D == row sum.
        c = __builtin_amdgcn_wmma_f32_16x16x32_f16(
            /*neg_a=*/false, a, /*neg_b=*/false, ones,
            /*c_mod=*/(short)0, c, /*reuse_a=*/false, /*reuse_b=*/false);
    }

    // lane 0 holds rows 0..7 (N=0), lane 16 holds rows 8..15 in c[0..7]
    if (l16 == 0) {
#pragma unroll
        for (int r = 0; r < 8; ++r)
            atomicAdd(&s_part[h * 8 + r], c[r]);
    }
    __syncthreads();

    if (tid < ROWS_PER_BLOCK) {
        const int o = rowBase + tid;
        const float bb  = (bias[o] >= s_rng[idxb[o] & 255]) ? 1.0f : 0.0f;
        const float tot = acc[o] + s_part[tid] + bb;
        out[o] = (tot >= bound[0]) ? 1.0f : 0.0f;
    }
}

extern "C" void kernel_launch(void* const* d_in, const int* in_sizes, int n_in,
                              void* d_out, int out_size, void* d_ws, size_t ws_size,
                              hipStream_t stream) {
    (void)in_sizes; (void)n_in; (void)out_size; (void)d_ws; (void)ws_size;
    const float* x      = (const float*)d_in[0];
    const float* wght   = (const float*)d_in[1];
    const float* bias   = (const float*)d_in[2];
    const float* rng    = (const float*)d_in[3];
    const float* acc    = (const float*)d_in[4];
    const float* bound  = (const float*)d_in[5];
    const int*   idxw   = (const int*)d_in[6];
    const int*   idxb   = (const int*)d_in[7];
    const int*   idxinv = (const int*)d_in[8];
    float* out = (float*)d_out;

    dim3 grid(OUT_F / ROWS_PER_BLOCK);   // 256 blocks
    dim3 block(THREADS);                 // 512 threads = 16 waves
    unary_linear_wmma<<<grid, block, 0, stream>>>(
        x, wght, bias, rng, acc, bound, idxw, idxb, idxinv, out);
}
